// SWT3_66030827209376
// MI455X (gfx1250) — matrix operand order repaired and checked by reading, not compiled, not run on hardware
//
#include <hip/hip_runtime.h>
#include <hip/hip_bf16.h>
#include <math.h>

typedef __attribute__((ext_vector_type(16))) __bf16 v16bf;
typedef __attribute__((ext_vector_type(8)))  float  v8f;

constexpr int BB   = 16;       // batch
constexpr int TT   = 12544;    // tokens per batch
constexpr int NTOK = BB * TT;  // 200704
constexpr int SPL  = 28;       // split-K partitions per batch
constexpr int CHT  = 32;       // tokens per chunk in reduction
constexpr int NCHUNK = TT / (SPL * CHT); // 14
constexpr float INV_SQRT_M = 0.17677669529663687f; // 1/sqrt(32)
constexpr float EPS_LN   = 1e-5f;
constexpr float EPS_ATTN = 1e-8f;

// ---- WMMA fragment helpers (layouts per CDNA5 ISA 7.12.2, wave32) ----
__device__ __forceinline__ v16bf load_afrag(const __bf16* p, int ld, int lane, int kofs) {
  int row = lane & 15, half = lane >> 4;
  v16bf a;
#pragma unroll
  for (int e = 0; e < 16; ++e) {
    int K = ((e & 8) ? 16 : 0) + half * 8 + (e & 7) + kofs;
    a[e] = p[row * ld + K];
  }
  return a;
}
// source layout: pn[n*ld + k] (per output column n, K contiguous)
__device__ __forceinline__ v16bf load_bfrag(const __bf16* pn, int ld, int lane, int kofs) {
  int n = lane & 15, half = lane >> 4;
  v16bf b;
#pragma unroll
  for (int e = 0; e < 16; ++e) {
    int K = ((e & 8) ? 16 : 0) + half * 8 + (e & 7) + kofs;
    b[e] = pn[n * ld + K];
  }
  return b;
}
__device__ __forceinline__ v8f wmma_bf16(v16bf a, v16bf b, v8f c) {
  return __builtin_amdgcn_wmma_f32_16x16x32_bf16(false, a, false, b, (short)0, c, false, false);
}

// ---- Kernel 0: unfold(4,2,1) + proj(16->64) + LN + PE ----
__global__ __launch_bounds__(256) void k_patch_embed(
    const float* __restrict__ X, const float* __restrict__ PW,
    const float* __restrict__ PB, const float* __restrict__ LG,
    const float* __restrict__ LB, const float* __restrict__ PE,
    float* __restrict__ H)
{
  __shared__ float sPW[16*64];
  __shared__ float sPB[64], sLG[64], sLB[64];
  int tid = threadIdx.x;
  for (int i = tid; i < 16*64; i += 256) sPW[i] = PW[i];
  if (tid < 64) { sPB[tid]=PB[tid]; sLG[tid]=LG[tid]; sLB[tid]=LB[tid]; }
  __syncthreads();
  int p = blockIdx.x * 256 + tid;
  int b = p / TT, t = p % TT;
  int py = t / 112, px = t % 112;
  float acc[64];
#pragma unroll
  for (int o = 0; o < 64; ++o) acc[o] = sPB[o];
  const float* xb = X + (size_t)b * 224 * 224;
#pragma unroll
  for (int ky = 0; ky < 4; ++ky) {
    int iy = 2*py - 1 + ky;
    if (iy < 0 || iy >= 224) continue;
#pragma unroll
    for (int kx = 0; kx < 4; ++kx) {
      int ix = 2*px - 1 + kx;
      if (ix < 0 || ix >= 224) continue;
      float v = xb[iy*224 + ix];
      const float* wr = &sPW[(ky*4+kx)*64];
#pragma unroll
      for (int o = 0; o < 64; ++o) acc[o] += v * wr[o];
    }
  }
  float mu = 0.f;
#pragma unroll
  for (int o = 0; o < 64; ++o) mu += acc[o];
  mu *= (1.f/64.f);
  float var = 0.f;
#pragma unroll
  for (int o = 0; o < 64; ++o) { float d = acc[o]-mu; var += d*d; }
  var *= (1.f/64.f);
  float rs = rsqrtf(var + EPS_LN);
  const float* pe = PE + (size_t)t * 64;
  float* out = H + (size_t)p * 64;
#pragma unroll
  for (int o = 0; o < 64; ++o)
    out[o] = (acc[o]-mu)*rs*sLG[o] + sLB[o] + pe[o];
}

// ---- Kernel A: LN1 -> kqv GEMM -> FAVOR+ features kp/qp (per 16-token tile) ----
__global__ __launch_bounds__(128) void k_front(
    const float* __restrict__ X,
    const float* __restrict__ G1, const float* __restrict__ B1,
    const float* __restrict__ KQVW, const float* __restrict__ KQVB,
    const float* __restrict__ WMAT,        // (32,64)
    float* __restrict__ V,                 // (N,64)
    __bf16* __restrict__ KP,               // (N,32)
    __bf16* __restrict__ QP)
{
  __shared__ __bf16 sWT[192*64];           // kqv_w transposed: [col][d]
  __shared__ __bf16 sWw[32*64];            // w: [m][d]
  __shared__ float  sKB[192];
  __shared__ float  sG[64], sB[64];
  __shared__ __bf16 sH[4][16*64];
  __shared__ __bf16 sK[4][16*64];
  __shared__ __bf16 sQ[4][16*64];
  __shared__ float  sXd[2][4][16];

  int tid = threadIdx.x, lane = tid & 31, wv = tid >> 5;
  for (int i = tid; i < 192*64; i += 128) {
    int col = i >> 6, d = i & 63;
    sWT[i] = (__bf16)KQVW[d*192 + col];
  }
  for (int i = tid; i < 32*64; i += 128) sWw[i] = (__bf16)WMAT[i];
  for (int i = tid; i < 192; i += 128) sKB[i] = KQVB[i];
  if (tid < 64) { sG[tid]=G1[tid]; sB[tid]=B1[tid]; }
  __syncthreads();

  int bt0 = blockIdx.x*64 + wv*16;
  { // LN1 -> sH (bf16), two lanes per token
    int tok = lane >> 1, part = lane & 1;
    const float* xp = X + (size_t)(bt0+tok)*64 + part*32;
    float s=0.f, ss=0.f, xv[32];
#pragma unroll
    for (int i=0;i<32;++i){ xv[i]=xp[i]; s+=xv[i]; ss+=xv[i]*xv[i]; }
    s += __shfl_xor(s,1,32); ss += __shfl_xor(ss,1,32);
    float mu = s*(1.f/64.f);
    float rs = rsqrtf(ss*(1.f/64.f) - mu*mu + EPS_LN);
#pragma unroll
    for (int i=0;i<32;++i){
      int d = part*32+i;
      sH[wv][tok*64+d] = (__bf16)((xv[i]-mu)*rs*sG[d]+sB[d]);
    }
  }
  __syncthreads();

  v16bf a0 = load_afrag(sH[wv], 64, lane, 0);
  v16bf a1 = load_afrag(sH[wv], 64, lane, 32);
  for (int j = 0; j < 12; ++j) {            // kqv: 12 column tiles of 16
    v8f c = {};
    v16bf b0 = load_bfrag(sWT + (j*16)*64, 64, lane, 0);
    v16bf b1 = load_bfrag(sWT + (j*16)*64, 64, lane, 32);
    c = wmma_bf16(a0,b0,c);
    c = wmma_bf16(a1,b1,c);
    int n = lane & 15, half = lane >> 4;
    float bias = sKB[j*16+n];
    if (j >= 8) {                           // v -> global fp32 (skip path)
      int col = (j-8)*16 + n;
#pragma unroll
      for (int r=0;r<8;++r){
        int tok = r + half*8;
        V[(size_t)(bt0+tok)*64 + col] = c[r] + bias;
      }
    } else {                                // k,q -> LDS bf16
      __bf16* dst = (j < 4) ? sK[wv] : sQ[wv];
      int col = (j & 3)*16 + n;
#pragma unroll
      for (int r=0;r<8;++r){
        int tok = r + half*8;
        dst[tok*64+col] = (__bf16)(c[r]+bias);
      }
    }
  }
  __syncthreads();
  { // xd = 0.5*||z||^2 per token
    int tok = lane>>1, part = lane&1;
    float sa=0.f, sb=0.f;
#pragma unroll
    for (int i=0;i<32;++i){
      float kv = (float)sK[wv][tok*64+part*32+i];
      float qv = (float)sQ[wv][tok*64+part*32+i];
      sa += kv*kv; sb += qv*qv;
    }
    sa += __shfl_xor(sa,1,32); sb += __shfl_xor(sb,1,32);
    if (part==0){ sXd[0][wv][tok]=0.5f*sa; sXd[1][wv][tok]=0.5f*sb; }
  }
  __syncthreads();
  for (int which = 0; which < 2; ++which) { // kp then qp
    const __bf16* zt = which ? sQ[wv] : sK[wv];
    __bf16* out = which ? QP : KP;
    v16bf za0 = load_afrag(zt, 64, lane, 0);
    v16bf za1 = load_afrag(zt, 64, lane, 32);
    for (int jm=0;jm<2;++jm){
      v8f c = {};
      v16bf b0 = load_bfrag(sWw + (jm*16)*64, 64, lane, 0);
      v16bf b1 = load_bfrag(sWw + (jm*16)*64, 64, lane, 32);
      c = wmma_bf16(za0,b0,c);
      c = wmma_bf16(za1,b1,c);
      int n = lane&15, half=lane>>4;
      int m = jm*16+n;
#pragma unroll
      for (int r=0;r<8;++r){
        int tok = r + half*8;
        float val = expf(c[r] - sXd[which][wv][tok]) * INV_SQRT_M;
        out[(size_t)(bt0+tok)*32 + m] = (__bf16)val;
      }
    }
  }
}

// ---- Kernel B1: split-K partials of kptv = v^T @ kp and ksum ----
__global__ __launch_bounds__(256) void k_kvred(
    const float* __restrict__ V, const __bf16* __restrict__ KP,
    float* __restrict__ PART)
{
  __shared__ __bf16 sV[64*34];    // [d][t]
  __shared__ __bf16 sKp[32*34];   // [m][t]
  int tid = threadIdx.x, lane = tid&31, wv = tid>>5;
  int b = blockIdx.x / SPL, s = blockIdx.x % SPL;
  size_t base = (size_t)b*TT + (size_t)s*(TT/SPL);
  int ntile = wv >> 1, mtile = wv & 1;   // 8 waves -> 4x2 C tiles of 16x16
  v8f c = {};
  float ks = 0.f;
  for (int ch = 0; ch < NCHUNK; ++ch) {
    size_t bt = base + (size_t)ch*CHT;
    for (int i = tid; i < 32*64; i += 256) {
      int t = i >> 6, d = i & 63;
      sV[d*34 + t] = (__bf16)V[(bt+t)*64 + d];
    }
    for (int i = tid; i < 32*32; i += 256) {
      int t = i >> 5, m = i & 31;
      sKp[m*34 + t] = KP[(bt+t)*32 + m];
    }
    __syncthreads();
    v16bf a  = load_afrag(sV  + ntile*16*34, 34, lane, 0);
    v16bf bb = load_bfrag(sKp + mtile*16*34, 34, lane, 0);
    c = wmma_bf16(a, bb, c);
    if (tid < 32) {
#pragma unroll
      for (int t=0;t<32;++t) ks += (float)sKp[tid*34 + t];
    }
    __syncthreads();
  }
  int n = lane&15, half = lane>>4;
  float* pp = PART + (size_t)blockIdx.x * 2080;
#pragma unroll
  for (int r=0;r<8;++r){
    int row = ntile*16 + r + half*8;
    pp[row*32 + mtile*16 + n] = c[r];
  }
  if (tid < 32) pp[2048 + tid] = ks;
}

// ---- Kernel B2: deterministic partial reduce -> kptv (bf16), ksum (f32) ----
__global__ __launch_bounds__(256) void k_kvfin(
    const float* __restrict__ PART, __bf16* __restrict__ KPTV,
    float* __restrict__ KSUM)
{
  int b = blockIdx.x;
  for (int i = threadIdx.x; i < 2080; i += 256) {
    float s = 0.f;
    const float* p = PART + (size_t)b*SPL*2080 + i;
    for (int j=0;j<SPL;++j) s += p[(size_t)j*2080];
    if (i < 2048) KPTV[(size_t)b*2048 + i] = (__bf16)s;
    else          KSUM[b*32 + (i-2048)] = s;
  }
}

// ---- Kernel C: attn/D -> proj+skip -> LN2 -> MLP(GELU) -> out ----
__global__ __launch_bounds__(128) void k_back(
    const __bf16* __restrict__ QP, const float* __restrict__ V,
    const __bf16* __restrict__ KPTV, const float* __restrict__ KSUM,
    const float* __restrict__ PJW, const float* __restrict__ PJB,
    const float* __restrict__ G2, const float* __restrict__ B2v,
    const float* __restrict__ M1W, const float* __restrict__ M1B,
    const float* __restrict__ M2W, const float* __restrict__ M2B,
    float* __restrict__ OUT)
{
  __shared__ __bf16 sPJ[64*64], sM1[64*64], sM2[64*64]; // transposed [n][k]
  __shared__ __bf16 sKT[64*32];                         // kptv [n][m]
  __shared__ float sKS[32];
  __shared__ float sPB[64], s1B[64], s2B[64], sG[64], sB[64];
  __shared__ float  sY[4][16*64];
  __shared__ __bf16 sT[4][16*64];
  __shared__ float  sD[4][16];
  int tid = threadIdx.x, lane = tid&31, wv = tid>>5;
  int bt0 = blockIdx.x*64 + wv*16;
  int b = (blockIdx.x*64) / TT;
  for (int i = tid; i < 64*64; i += 128) {
    int n = i >> 6, k = i & 63;
    sPJ[i] = (__bf16)PJW[k*64+n];
    sM1[i] = (__bf16)M1W[k*64+n];
    sM2[i] = (__bf16)M2W[k*64+n];
  }
  for (int i = tid; i < 2048; i += 128) sKT[i] = KPTV[(size_t)b*2048 + i];
  if (tid < 64) { sPB[tid]=PJB[tid]; s1B[tid]=M1B[tid]; s2B[tid]=M2B[tid];
                  sG[tid]=G2[tid]; sB[tid]=B2v[tid]; }
  if (tid < 32) sKS[tid] = KSUM[b*32+tid];
  __syncthreads();

  { // D = qp . ksum
    int tok = lane>>1, part = lane&1;
    const __bf16* qp = QP + (size_t)(bt0+tok)*32 + part*16;
    float d = 0.f;
#pragma unroll
    for (int i=0;i<16;++i) d += (float)qp[i] * sKS[part*16+i];
    d += __shfl_xor(d,1,32);
    if (part==0) sD[wv][tok] = d;
  }
  __syncthreads();
  { // y_raw = qp @ kptv^T, /(D+eps) -> sT
    v16bf a = load_afrag(QP + (size_t)bt0*32, 32, lane, 0);
    int n = lane&15, half = lane>>4;
    for (int jn=0;jn<4;++jn){
      v8f c = {};
      v16bf bb = load_bfrag(sKT + (jn*16)*32, 32, lane, 0);
      c = wmma_bf16(a,bb,c);
#pragma unroll
      for (int r=0;r<8;++r){
        int tok = r + half*8;
        sT[wv][tok*64 + jn*16 + n] = (__bf16)(c[r] / (sD[wv][tok] + EPS_ATTN));
      }
    }
  }
  __syncthreads();
  { // proj + bias + v skip -> sY (f32)
    v16bf a0 = load_afrag(sT[wv], 64, lane, 0);
    v16bf a1 = load_afrag(sT[wv], 64, lane, 32);
    int n = lane&15, half = lane>>4;
    for (int jn=0;jn<4;++jn){
      v8f c = {};
      v16bf b0 = load_bfrag(sPJ + (jn*16)*64, 64, lane, 0);
      v16bf b1 = load_bfrag(sPJ + (jn*16)*64, 64, lane, 32);
      c = wmma_bf16(a0,b0,c); c = wmma_bf16(a1,b1,c);
      int col = jn*16+n;
#pragma unroll
      for (int r=0;r<8;++r){
        int tok = r + half*8;
        sY[wv][tok*64+col] = c[r] + sPB[col] + V[(size_t)(bt0+tok)*64+col];
      }
    }
  }
  __syncthreads();
  { // LN2 -> sT
    int tok = lane>>1, part = lane&1;
    float s=0.f, ss=0.f, xv[32];
#pragma unroll
    for (int i=0;i<32;++i){ xv[i]=sY[wv][tok*64+part*32+i]; s+=xv[i]; ss+=xv[i]*xv[i]; }
    s += __shfl_xor(s,1,32); ss += __shfl_xor(ss,1,32);
    float mu = s*(1.f/64.f);
    float rs = rsqrtf(ss*(1.f/64.f)-mu*mu+EPS_LN);
#pragma unroll
    for (int i=0;i<32;++i){
      int d = part*32+i;
      sT[wv][tok*64+d] = (__bf16)((xv[i]-mu)*rs*sG[d]+sB[d]);
    }
  }
  __syncthreads();
  { // mlp1 + exact GELU -> sT (in place; frags loaded first)
    v16bf a0 = load_afrag(sT[wv], 64, lane, 0);
    v16bf a1 = load_afrag(sT[wv], 64, lane, 32);
    int n = lane&15, half = lane>>4;
    for (int jn=0;jn<4;++jn){
      v8f c = {};
      v16bf b0 = load_bfrag(sM1 + (jn*16)*64, 64, lane, 0);
      v16bf b1 = load_bfrag(sM1 + (jn*16)*64, 64, lane, 32);
      c = wmma_bf16(a0,b0,c); c = wmma_bf16(a1,b1,c);
      int col = jn*16+n;
#pragma unroll
      for (int r=0;r<8;++r){
        int tok = r+half*8;
        float h = c[r] + s1B[col];
        sT[wv][tok*64+col] = (__bf16)(0.5f*h*(1.f+erff(h*0.70710678118654752f)));
      }
    }
  }
  __syncthreads();
  { // mlp2 + bias + residual(y) -> OUT
    v16bf a0 = load_afrag(sT[wv], 64, lane, 0);
    v16bf a1 = load_afrag(sT[wv], 64, lane, 32);
    int n = lane&15, half = lane>>4;
    for (int jn=0;jn<4;++jn){
      v8f c = {};
      v16bf b0 = load_bfrag(sM2 + (jn*16)*64, 64, lane, 0);
      v16bf b1 = load_bfrag(sM2 + (jn*16)*64, 64, lane, 32);
      c = wmma_bf16(a0,b0,c); c = wmma_bf16(a1,b1,c);
      int col = jn*16+n;
#pragma unroll
      for (int r=0;r<8;++r){
        int tok = r+half*8;
        OUT[(size_t)(bt0+tok)*64+col] = c[r] + s2B[col] + sY[wv][tok*64+col];
      }
    }
  }
}

extern "C" void kernel_launch(void* const* d_in, const int* in_sizes, int n_in,
                              void* d_out, int out_size, void* d_ws, size_t ws_size,
                              hipStream_t stream)
{
  (void)in_sizes; (void)n_in; (void)out_size; (void)ws_size;
  const float* X  = (const float*)d_in[0];
  const float* PW = (const float*)d_in[1];
  const float* PB = (const float*)d_in[2];
  const float* LG = (const float*)d_in[3];
  const float* LB = (const float*)d_in[4];
  const float* PE = (const float*)d_in[5];

  // Workspace layout (~132 MB total)
  char* ws = (char*)d_ws;
  size_t off = 0;
  auto alloc = [&](size_t bytes){ void* p = ws + off; off += (bytes + 255) & ~(size_t)255; return p; };
  float*  gH    = (float*) alloc((size_t)NTOK*64*4);
  float*  gV    = (float*) alloc((size_t)NTOK*64*4);
  __bf16* gKP   = (__bf16*)alloc((size_t)NTOK*32*2);
  __bf16* gQP   = (__bf16*)alloc((size_t)NTOK*32*2);
  float*  gPart = (float*) alloc((size_t)BB*SPL*2080*4);
  __bf16* gKPTV = (__bf16*)alloc((size_t)BB*2048*2);
  float*  gKSUM = (float*) alloc((size_t)BB*32*4);

  k_patch_embed<<<NTOK/256, 256, 0, stream>>>(X, PW, PB, LG, LB, PE, gH);

  for (int blk = 0; blk < 2; ++blk) {
    int base = 6 + blk*13;  // p dict order: n1g n1b kqv_w kqv_b proj_w proj_b n2g n2b m1w m1b m2w m2b w
    const float* n1g = (const float*)d_in[base+0];
    const float* n1b = (const float*)d_in[base+1];
    const float* kqvw= (const float*)d_in[base+2];
    const float* kqvb= (const float*)d_in[base+3];
    const float* pjw = (const float*)d_in[base+4];
    const float* pjb = (const float*)d_in[base+5];
    const float* n2g = (const float*)d_in[base+6];
    const float* n2b = (const float*)d_in[base+7];
    const float* m1w = (const float*)d_in[base+8];
    const float* m1b = (const float*)d_in[base+9];
    const float* m2w = (const float*)d_in[base+10];
    const float* m2b = (const float*)d_in[base+11];
    const float* wm  = (const float*)d_in[base+12];

    float* outp = (blk == 1) ? (float*)d_out : gH;  // block0 writes h in place
    k_front<<<NTOK/64, 128, 0, stream>>>(gH, n1g, n1b, kqvw, kqvb, wm, gV, gKP, gQP);
    k_kvred<<<BB*SPL, 256, 0, stream>>>(gV, gKP, gPart);
    k_kvfin<<<BB, 256, 0, stream>>>(gPart, gKPTV, gKSUM);
    k_back<<<NTOK/64, 128, 0, stream>>>(gQP, gV, gKPTV, gKSUM, pjw, pjb,
                                        n2g, n2b, m1w, m1b, m2w, m2b, outp);
  }
}
